// TransformerDecoder_44272522887860
// MI455X (gfx1250) — matrix operand order repaired and checked by reading, not compiled, not run on hardware
//
#include <hip/hip_runtime.h>
#include <hip/hip_bf16.h>

// ---------------------------------------------------------------------------
// Transformer decoder fwd for MI455X (gfx1250, wave32, WMMA).
// All GEMMs use v_wmma_f32_16x16x32_f16 (K=32 == H, fp32 accumulate).
// Attention core (DH=4) is a fused 2-pass online-softmax VALU kernel.
// ---------------------------------------------------------------------------

#define L_    6
#define B_    32
#define S_    128
#define H_    32
#define NH_   8
#define DH_   4
#define V_    32000
#define FFH_  128
#define NTOK_ (B_ * S_)        // 4096
#define NEGB  (-1.0e9f)

typedef __attribute__((ext_vector_type(16))) _Float16 v16h;
typedef __attribute__((ext_vector_type(8)))  float    v8f;

// ---- WMMA tile loaders (layouts per CDNA5 ISA §7.12.2) ----------

// A: 16x32 tile of row-major [M,K] fp32 matrix, converted to f16.
// lane m = lane&15, half = lane>>4 ; element e -> K = (e>=8?16:0)+half*8+((e>>1)&3)*2+(e&1)
__device__ __forceinline__ v16h load_a_tile(const float* __restrict__ A, int lda,
                                            int row0, int k0, int lane) {
  const int m = lane & 15, half = lane >> 4;
  const float* base = A + (size_t)(row0 + m) * lda + k0;
  v16h a;
#pragma unroll
  for (int e = 0; e < 16; ++e) {
    const int i = e >> 1, j = e & 1;
    const int kk = ((i >= 4) ? 16 : 0) + half * 8 + (i & 3) * 2 + j;
    a[e] = (_Float16)base[kk];
  }
  return a;
}

// B: 32x16 tile of row-major [K,N] fp32 matrix, converted to f16.
// lane n = lane&15, half = lane>>4 ; element e -> K = (e>=8?16:0)+half*8+(e&7)
__device__ __forceinline__ v16h load_b_tile(const float* __restrict__ W, int ldb,
                                            int k0, int col0, int lane) {
  const int n = lane & 15, half = lane >> 4;
  const float* base = W + (size_t)(k0 + half * 8) * ldb + col0 + n;
  v16h b;
#pragma unroll
  for (int e = 0; e < 16; ++e) b[e] = (_Float16)base[(size_t)((e >= 8) ? e + 8 : e) * ldb];
  return b;
}

// ---- Generic GEMM: C[M,N] = A[M,KDIM] @ W[KDIM,N] + bias, optional ReLU ----
template <int KDIM, bool RELU>
__global__ void gemm_bias_kernel(const float* __restrict__ A,
                                 const float* __restrict__ W,
                                 const float* __restrict__ bias,
                                 float* __restrict__ C,
                                 int M, int N, int total_tiles) {
  const int lane = threadIdx.x & 31;
  const int wave = threadIdx.x >> 5;
  const int tile = blockIdx.x * (blockDim.x >> 5) + wave;
  if (tile >= total_tiles) return;                 // wave-uniform: EXEC stays full
  const int ntn = N >> 4;
  const int mt = tile / ntn, nt = tile % ntn;

  v8f c = {};
#pragma unroll
  for (int k0 = 0; k0 < KDIM; k0 += 32) {
    v16h a = load_a_tile(A, KDIM, mt * 16, k0, lane);
    v16h b = load_b_tile(W, N, k0, nt * 16, lane);
    c = __builtin_amdgcn_wmma_f32_16x16x32_f16(false, a, false, b,
                                               (short)0, c, false, false);
  }
  const int n = lane & 15, half = lane >> 4;
  const float bv = bias[nt * 16 + n];
#pragma unroll
  for (int r = 0; r < 8; ++r) {
    float v = c[r] + bv;
    if (RELU) v = fmaxf(v, 0.0f);
    C[(size_t)(mt * 16 + r + 8 * half) * N + nt * 16 + n] = v;
  }
}

// ---- Embedding gather: x[t,f] = emb[dec[t], f] -----------------------------
__global__ void embed_kernel(const float* __restrict__ emb,
                             const int* __restrict__ dec,
                             float* __restrict__ x) {
  const int idx = blockIdx.x * blockDim.x + threadIdx.x;   // token*H + f
  const int t = idx >> 5, f = idx & 31;
  x[idx] = emb[(size_t)dec[t] * H_ + f];
}

// ---- Fused attention core: scores (+relpos) + mask + softmax + att@V -------
// grid = B*NH blocks, 128 threads (one q row per thread), 2-pass softmax.
__global__ void attn_core_kernel(const float* __restrict__ Q,
                                 const float* __restrict__ K,
                                 const float* __restrict__ V,
                                 const float* __restrict__ pos,      // [2S-1,DH] or null
                                 const int* __restrict__ mask_tok,   // pad ids over keys
                                 int causal,
                                 float* __restrict__ O) {
  const int b = blockIdx.x / NH_, h = blockIdx.x % NH_;
  const int tq = threadIdx.x;   // 0..S-1

  __shared__ float ks[S_][DH_];
  __shared__ float vs[S_][DH_];
  __shared__ float padm[S_];

#pragma unroll
  for (int d = 0; d < DH_; ++d) {
    ks[tq][d] = K[(size_t)(b * S_ + tq) * H_ + h * DH_ + d];
    vs[tq][d] = V[(size_t)(b * S_ + tq) * H_ + h * DH_ + d];
  }
  padm[tq] = (mask_tok[b * S_ + tq] == 2) ? NEGB : 0.0f;
  __syncthreads();

  float qv[DH_];
#pragma unroll
  for (int d = 0; d < DH_; ++d)
    qv[d] = Q[(size_t)(b * S_ + tq) * H_ + h * DH_ + d] * 0.5f;  // 1/sqrt(DH)

  // pass 1: row max
  float mx = -3.0e38f;
  for (int kk = 0; kk < S_; ++kk) {
    float s = 0.0f;
#pragma unroll
    for (int d = 0; d < DH_; ++d) s += qv[d] * ks[kk][d];
    if (pos) {
      const float* re = pos + (size_t)(kk - tq + (S_ - 1)) * DH_;
#pragma unroll
      for (int d = 0; d < DH_; ++d) s += qv[d] * re[d];
    }
    s += padm[kk];
    if (causal && kk > tq) s += NEGB;
    mx = fmaxf(mx, s);
  }
  // pass 2: exp-sum and weighted V accumulate
  float sum = 0.0f, acc[DH_] = {0.0f, 0.0f, 0.0f, 0.0f};
  for (int kk = 0; kk < S_; ++kk) {
    float s = 0.0f;
#pragma unroll
    for (int d = 0; d < DH_; ++d) s += qv[d] * ks[kk][d];
    if (pos) {
      const float* re = pos + (size_t)(kk - tq + (S_ - 1)) * DH_;
#pragma unroll
      for (int d = 0; d < DH_; ++d) s += qv[d] * re[d];
    }
    s += padm[kk];
    if (causal && kk > tq) s += NEGB;
    const float e = expf(s - mx);
    sum += e;
#pragma unroll
    for (int d = 0; d < DH_; ++d) acc[d] += e * vs[kk][d];
  }
  const float inv = 1.0f / sum;
#pragma unroll
  for (int d = 0; d < DH_; ++d)
    O[(size_t)(b * S_ + tq) * H_ + h * DH_ + d] = acc[d] * inv;
}

// ---- Residual add + LayerNorm over H=32 (one thread per token) -------------
__global__ void add_ln_kernel(const float* __restrict__ a,
                              const float* __restrict__ x,
                              const float* __restrict__ g,
                              const float* __restrict__ be,
                              float* __restrict__ out) {
  const int t = blockIdx.x * blockDim.x + threadIdx.x;
  if (t >= NTOK_) return;
  float v[H_];
  float m = 0.0f;
#pragma unroll
  for (int i = 0; i < H_; ++i) { v[i] = a[(size_t)t * H_ + i] + x[(size_t)t * H_ + i]; m += v[i]; }
  m *= (1.0f / H_);
  float var = 0.0f;
#pragma unroll
  for (int i = 0; i < H_; ++i) { const float d = v[i] - m; var += d * d; }
  var *= (1.0f / H_);
  const float r = rsqrtf(var + 1e-5f);
#pragma unroll
  for (int i = 0; i < H_; ++i)
    out[(size_t)t * H_ + i] = (v[i] - m) * r * g[i] + be[i];
}

// ---------------------------------------------------------------------------
extern "C" void kernel_launch(void* const* d_in, const int* in_sizes, int n_in,
                              void* d_out, int out_size, void* d_ws, size_t ws_size,
                              hipStream_t stream) {
  const float* emb     = (const float*)d_in[0];
  const float* self_w  = (const float*)d_in[1];
  const float* self_b  = (const float*)d_in[2];
  const float* self_pos= (const float*)d_in[3];
  const float* cross_w = (const float*)d_in[4];
  const float* cross_b = (const float*)d_in[5];
  const float* ln_g    = (const float*)d_in[6];
  const float* ln_b    = (const float*)d_in[7];
  const float* ff1_w   = (const float*)d_in[8];
  const float* ff1_b   = (const float*)d_in[9];
  const float* ff2_w   = (const float*)d_in[10];
  const float* ff2_b   = (const float*)d_in[11];
  const float* out_w   = (const float*)d_in[12];
  const float* out_b   = (const float*)d_in[13];
  const float* enc     = (const float*)d_in[14];
  const int*   dec_in  = (const int*)d_in[15];
  const int*   enc_in  = (const int*)d_in[16];
  float* logits = (float*)d_out;

  // workspace carve-up (fp32)
  float* ws = (float*)d_ws;
  const size_t TH = (size_t)NTOK_ * H_;        // 131072
  float* X0 = ws;            // hidden state (layer input / final LN out)
  float* X1 = X0 + TH;       // after self-attn block
  float* X2 = X1 + TH;       // after cross-attn block
  float* Qb = X2 + TH;
  float* Kb = Qb + TH;
  float* Vb = Kb + TH;
  float* Ab = Vb + TH;       // attention core output
  float* Pb = Ab + TH;       // projection / ff2 output
  float* FF = Pb + TH;       // [NTOK, 4H]

  #define GEMM(KD, RELU, A, W, Bi, C, M, N)                                     \
    do {                                                                        \
      const int tiles = ((M) / 16) * ((N) / 16);                                \
      gemm_bias_kernel<KD, RELU><<<(tiles + 7) / 8, 256, 0, stream>>>(          \
          (A), (W), (Bi), (C), (M), (N), tiles);                                \
    } while (0)

  // x = embedding[decoder_input]
  embed_kernel<<<(NTOK_ * H_) / 256, 256, 0, stream>>>(emb, dec_in, X0);

  for (int l = 0; l < L_; ++l) {
    const float* sw  = self_w  + (size_t)l * 4 * H_ * H_;
    const float* sb  = self_b  + (size_t)l * 4 * H_;
    const float* sp  = self_pos+ (size_t)l * (2 * S_ - 1) * DH_;
    const float* cw  = cross_w + (size_t)l * 4 * H_ * H_;
    const float* cb  = cross_b + (size_t)l * 4 * H_;
    const float* lg  = ln_g    + (size_t)l * 3 * H_;
    const float* lb  = ln_b    + (size_t)l * 3 * H_;

    // ---- self attention ----
    GEMM(32, false, X0, sw + 0 * H_ * H_, sb + 0 * H_, Qb, NTOK_, H_);
    GEMM(32, false, X0, sw + 1 * H_ * H_, sb + 1 * H_, Kb, NTOK_, H_);
    GEMM(32, false, X0, sw + 2 * H_ * H_, sb + 2 * H_, Vb, NTOK_, H_);
    attn_core_kernel<<<B_ * NH_, S_, 0, stream>>>(Qb, Kb, Vb, sp, dec_in, 1, Ab);
    GEMM(32, false, Ab, sw + 3 * H_ * H_, sb + 3 * H_, Pb, NTOK_, H_);
    add_ln_kernel<<<NTOK_ / 256, 256, 0, stream>>>(Pb, X0, lg + 0 * H_, lb + 0 * H_, X1);

    // ---- cross attention (K/V from encoder outputs) ----
    GEMM(32, false, X1,  cw + 0 * H_ * H_, cb + 0 * H_, Qb, NTOK_, H_);
    GEMM(32, false, enc, cw + 1 * H_ * H_, cb + 1 * H_, Kb, NTOK_, H_);
    GEMM(32, false, enc, cw + 2 * H_ * H_, cb + 2 * H_, Vb, NTOK_, H_);
    attn_core_kernel<<<B_ * NH_, S_, 0, stream>>>(Qb, Kb, Vb, nullptr, enc_in, 0, Ab);
    GEMM(32, false, Ab, cw + 3 * H_ * H_, cb + 3 * H_, Pb, NTOK_, H_);
    add_ln_kernel<<<NTOK_ / 256, 256, 0, stream>>>(Pb, X1, lg + 1 * H_, lb + 1 * H_, X2);

    // ---- feed-forward ----
    GEMM(32,  true,  X2, ff1_w + (size_t)l * H_ * FFH_,  ff1_b + (size_t)l * FFH_, FF, NTOK_, FFH_);
    GEMM(128, false, FF, ff2_w + (size_t)l * FFH_ * H_,  ff2_b + (size_t)l * H_,   Pb, NTOK_, H_);
    add_ln_kernel<<<NTOK_ / 256, 256, 0, stream>>>(Pb, X2, lg + 2 * H_, lb + 2 * H_, X0);
  }

  // final vocab projection: [4096,32] @ [32,32000] + bias  (store-BW bound)
  GEMM(32, false, X0, out_w, out_b, logits, NTOK_, V_);

  #undef GEMM
}
